// MultiScaleRetention_44633300140813
// MI455X (gfx1250) — matrix operand order repaired and checked by reading, not compiled, not run on hardware
//
#include <hip/hip_runtime.h>
#include <hip/hip_bf16.h>
#include <math.h>

#define BB      4
#define SEQ     2048
#define DIMC    1024
#define HD      128
#define NH      8

typedef __attribute__((ext_vector_type(16))) __bf16       v16bf;
typedef __attribute__((ext_vector_type(8)))  float        v8f;
typedef __attribute__((ext_vector_type(4)))  unsigned int v4u;
typedef __attribute__((ext_vector_type(8)))  unsigned int v8u;

union Frag { v16bf bf; v4u u[2]; };

__device__ __forceinline__ v8f v8f_zero() {
  v8f v;
#pragma unroll
  for (int i = 0; i < 8; ++i) v[i] = 0.f;
  return v;
}

// A-matrix 16x32 bf16 fragment (M rows of `base`, row stride `stride` elems).
// Lane L<16: row m0+L, K = {k0..k0+7} and {k0+16..k0+23}; lane L+16: K+8.
__device__ __forceinline__ void load_a_frag(const __bf16* base, int stride,
                                            int m0, int k0, int lane, Frag* f) {
  const __bf16* p = base + (size_t)(m0 + (lane & 15)) * stride + k0 + ((lane >> 4) << 3);
  f->u[0] = *(const v4u*)p;
  f->u[1] = *(const v4u*)(p + 16);
}

// B-matrix 32x16 bf16 fragment where operand is stored "N-major": row n holds
// the K values contiguously. Lane L<16: n0+L, K = {k0..k0+7} and {k0+16..k0+23}; lane L+16: K+8.
__device__ __forceinline__ void load_b_frag(const __bf16* base, int stride,
                                            int n0, int k0, int lane, Frag* f) {
  const __bf16* p = base + (size_t)(n0 + (lane & 15)) * stride + k0 + ((lane >> 4) << 3);
  f->u[0] = *(const v4u*)p;
  f->u[1] = *(const v4u*)(p + 16);
}

__device__ __forceinline__ v8f wmma_bf16(const Frag& a, const Frag& b, v8f c) {
  return __builtin_amdgcn_wmma_f32_16x16x32_bf16(false, a.bf, false, b.bf,
                                                 (short)0, c, false, false);
}

// ---------------------------------------------------------------- converters
__global__ __launch_bounds__(256) void cvt_bf16_kernel(const float* src, __bf16* dst, int n) {
  int i = blockIdx.x * 256 + threadIdx.x;
  if (i < n) dst[i] = (__bf16)src[i];
}

// dst[b][c][r] = src[b][r][c]  (transpose + f32->bf16)
__global__ __launch_bounds__(256) void transpose_cvt_kernel(const float* src, __bf16* dst,
                                                            int R, int C) {
  int c = blockIdx.x * 16 + (threadIdx.x & 15);
  int r = blockIdx.y * 16 + (threadIdx.x >> 4);
  size_t base = (size_t)blockIdx.z * R * C;
  if (r < R && c < C)
    dst[base + (size_t)c * R + r] = (__bf16)src[base + (size_t)r * C + c];
}

// ------------------------------------------------- X = qh @ W_h  (per head)
__global__ __launch_bounds__(256) void proj_kernel(const __bf16* qbf, const __bf16* WqkvT,
                                                   __bf16* Xbf, __bf16* XbfT) {
  int wave = threadIdx.x >> 5, lane = threadIdx.x & 31;
  int h = blockIdx.y, b = blockIdx.z;
  int m0 = blockIdx.x * 16;      // s-tile
  int n0 = wave * 16;            // output-dim tile (8 waves cover 128)
  const __bf16* A  = qbf + (size_t)b * SEQ * DIMC + h * HD;   // row stride DIMC
  const __bf16* BT = WqkvT + (size_t)h * HD * HD;             // [n][k]
  v8f acc = v8f_zero();
#pragma unroll
  for (int k0 = 0; k0 < HD; k0 += 32) {
    Frag a, bfr;
    load_a_frag(A, DIMC, m0, k0, lane, &a);
    load_b_frag(BT, HD, n0, k0, lane, &bfr);
    acc = wmma_bf16(a, bfr, acc);
  }
  __bf16* Xo  = Xbf  + (size_t)(b * NH + h) * SEQ * HD;
  __bf16* XoT = XbfT + (size_t)(b * NH + h) * HD * SEQ;
  int n = lane & 15, hi = lane >> 4;
#pragma unroll
  for (int r = 0; r < 8; ++r) {
    int m = r + 8 * hi;
    __bf16 v = (__bf16)acc[r];
    Xo [(size_t)(m0 + m) * HD  + n0 + n] = v;
    XoT[(size_t)(n0 + n) * SEQ + m0 + m] = v;
  }
}

// ------------------------------- pass 1: decay-weighted column sums (per b,h)
// Block owns 8 consecutive t-tiles; s-tiles of X are streamed into LDS ONCE
// per block by the Tensor Data Mover (8x traffic reduction vs per-wave loads).
__global__ __launch_bounds__(256) void pass1_colsum_kernel(const __bf16* Xbf, float* colsum) {
  extern __shared__ __bf16 sA[];   // 16x128 bf16 staging tile (4 KB), TDM-filled
  int wave = threadIdx.x >> 5, lane = threadIdx.x & 31;
  int h = blockIdx.y, b = blockIdx.z;
  int tBase = blockIdx.x * 128;
  int t0 = tBase + wave * 16;
  const __bf16* X = Xbf + (size_t)(b * NH + h) * SEQ * HD;

  float gamma = 1.f - exp2f(-5.f - (float)h);
  float lg = logf(gamma);
  int n = lane & 15, hi = lane >> 4;
  int t = t0 + n;
  // closed-form column sum of D:  sum_{s>=t} gamma^(s-t) = (1-gamma^(S-t))/(1-gamma)
  float colD   = (1.f - expf((float)(SEQ - t) * lg)) / (1.f - gamma);
  float cscale = rsqrtf(colD) * 0.08838834764831845f;   // 1/sqrt(HD)
  float gn = expf(-(float)n * lg);                      // gamma^(-n)
  float g8 = hi ? expf(8.f * lg) : 1.f;                 // gamma^(8*hi)

  Frag bfr[4];
#pragma unroll
  for (int kc = 0; kc < 4; ++kc) load_b_frag(X, HD, t0, kc * 32, lane, &bfr[kc]);

  float colacc = 0.f;
  for (int s0 = tBase; s0 < SEQ; s0 += 16) {   // block-uniform trip count
    if (wave == 0) {
      // D# group0: count=1, lds_addr=0 (dynamic-LDS base), global addr, type=2
      unsigned long long ga = (unsigned long long)(uintptr_t)(X + (size_t)s0 * HD);
      v4u g0;
      g0[0] = 1u;
      g0[1] = 0u;
      g0[2] = (unsigned)ga;
      g0[3] = ((unsigned)(ga >> 32) & 0x01FFFFFFu) | 0x80000000u;
      // D# group1: data_size=2B, tensor_dim0=128, tensor_dim1=16,
      // tile_dim0=128, tile_dim1=16, tensor_dim0_stride=128
      v8u g1;
      g1[0] = 1u << 16;        // data_size = 1 (2 bytes)
      g1[1] = 128u << 16;      // tensor_dim0[15:0] in high half
      g1[2] = 16u << 16;       // tensor_dim0[31:16]=0 | tensor_dim1 low
      g1[3] = 128u << 16;      // tensor_dim1 hi = 0 | tile_dim0 = 128
      g1[4] = 16u;             // tile_dim1 = 16, tile_dim2 = 0
      g1[5] = 128u;            // tensor_dim0_stride low 32 = 128
      g1[6] = 0u;              // stride hi | dim1_stride low
      g1[7] = 0u;
      asm volatile("tensor_load_to_lds %0, %1" :: "s"(g0), "s"(g1) : "memory");
      __builtin_amdgcn_s_wait_tensorcnt(0);
    }
    __syncthreads();
    if (t0 <= s0) {
      Frag a[4];
      const __bf16* rowp = sA + (size_t)(lane & 15) * HD + (hi << 3);
#pragma unroll
      for (int kc = 0; kc < 4; ++kc) {
        const __bf16* p = rowp + kc * 32;
        a[kc].u[0] = *(const v4u*)p;           // ds_load_b128
        a[kc].u[1] = *(const v4u*)(p + 16);
      }
      v8f acc = v8f_zero();
#pragma unroll
      for (int kc = 0; kc < 4; ++kc) acc = wmma_bf16(a[kc], bfr[kc], acc);

      float f = expf((float)(s0 - t0) * lg) * cscale;
      float gm = g8;
#pragma unroll
      for (int r = 0; r < 8; ++r) {
        int m = r + 8 * hi;
        float d = (s0 == t0) ? ((m >= n) ? f * gm * gn : 0.f) : f * gm * gn;
        colacc += acc[r] * d;
        gm *= gamma;
      }
    }
    __syncthreads();   // WAR: next TDM write must not overtake reads
  }
  colacc += __shfl_xor(colacc, 16, 32);
  if (lane < 16) colsum[(size_t)(b * NH + h) * SEQ + t] = colacc;
}

// --------------------------- pass 2: Y = (P/den) @ V  streamed over t-tiles
__global__ __launch_bounds__(256) void pass2_out_kernel(const __bf16* Xbf, const __bf16* XbfT,
                                                        const float* colsum, float* Y) {
  __shared__ float lds[8][16][33];   // per-wave 16x32 P tile (padded)
  int wave = threadIdx.x >> 5, lane = threadIdx.x & 31;
  int h = blockIdx.y, b = blockIdx.z;
  int s0 = (blockIdx.x * 8 + wave) * 16;
  const __bf16* X  = Xbf  + (size_t)(b * NH + h) * SEQ * HD;
  const __bf16* XT = XbfT + (size_t)(b * NH + h) * HD * SEQ;
  const float*  cs = colsum + (size_t)(b * NH + h) * SEQ;

  float gamma = 1.f - exp2f(-5.f - (float)h);
  float lg = logf(gamma);
  int n = lane & 15, hi = lane >> 4;
  float gn = expf(-(float)n * lg);
  float g8 = hi ? expf(8.f * lg) : 1.f;

  Frag a1[4];
#pragma unroll
  for (int kc = 0; kc < 4; ++kc) load_a_frag(X, HD, s0, kc * 32, lane, &a1[kc]);

  v8f out[8];
#pragma unroll
  for (int i = 0; i < 8; ++i) out[i] = v8f_zero();

  float (*P)[33] = lds[wave];

  for (int t0 = 0; t0 <= s0; t0 += 32) {
#pragma unroll
    for (int half = 0; half < 2; ++half) {
      int tt = t0 + half * 16;
      if (tt <= s0) {
        Frag bfr[4];
#pragma unroll
        for (int kc = 0; kc < 4; ++kc) load_b_frag(X, HD, tt, kc * 32, lane, &bfr[kc]);
        v8f acc = v8f_zero();
#pragma unroll
        for (int kc = 0; kc < 4; ++kc) acc = wmma_bf16(a1[kc], bfr[kc], acc);

        int t = tt + n;
        float colD = (1.f - expf((float)(SEQ - t) * lg)) / (1.f - gamma);
        float den  = fmaxf(fabsf(cs[t]), 1.f);
        float f    = expf((float)(s0 - tt) * lg) * rsqrtf(colD)
                     * 0.08838834764831845f / den;
        float gm = g8;
#pragma unroll
        for (int r = 0; r < 8; ++r) {
          int m = r + 8 * hi;
          float d = (tt == s0) ? ((m >= n) ? f * gm * gn : 0.f) : f * gm * gn;
          P[m][half * 16 + n] = acc[r] * d;
          gm *= gamma;
        }
      } else {
#pragma unroll
        for (int r = 0; r < 8; ++r) P[r + 8 * hi][half * 16 + n] = 0.f;
      }
    }
    asm volatile("s_wait_dscnt 0" ::: "memory");   // in-order LDS pipe: cross-lane RAW safe

    // transpose D-layout -> A-fragment (16x32 bf16)
    Frag a2;
    int row = lane & 15, kb = hi * 8;
#pragma unroll
    for (int j = 0; j < 4; ++j) {
      a2.bf[2 * j]     = (__bf16)P[row][kb + 2 * j];
      a2.bf[2 * j + 1] = (__bf16)P[row][kb + 2 * j + 1];
      a2.bf[8 + 2 * j]     = (__bf16)P[row][kb + 16 + 2 * j];
      a2.bf[8 + 2 * j + 1] = (__bf16)P[row][kb + 16 + 2 * j + 1];
    }
#pragma unroll
    for (int nc = 0; nc < 8; ++nc) {
      Frag b2;
      load_b_frag(XT, SEQ, nc * 16, t0, lane, &b2);  // contiguous via transposed X
      out[nc] = wmma_bf16(a2, b2, out[nc]);
    }
    asm volatile("s_wait_dscnt 0" ::: "memory");     // WAR before next tile's P writes
  }

#pragma unroll
  for (int nc = 0; nc < 8; ++nc)
#pragma unroll
    for (int r = 0; r < 8; ++r) {
      int m = r + 8 * hi;
      Y[((size_t)b * SEQ + s0 + m) * DIMC + h * HD + nc * 16 + n] = out[nc][r];
    }
}

// ------------------------------------------------------------- group norm
__global__ __launch_bounds__(1024) void groupnorm_kernel(const float* Y, const float* beta,
                                                         __bf16* Ygn) {
  size_t bs = blockIdx.x;
  int c = threadIdx.x;                 // wave w == group w (32 channels/group)
  float x = Y[bs * DIMC + c];
  float s1 = x, s2 = x * x;
#pragma unroll
  for (int off = 16; off; off >>= 1) {
    s1 += __shfl_xor(s1, off, 32);
    s2 += __shfl_xor(s2, off, 32);
  }
  float mu  = s1 * (1.f / 32.f);
  float var = s2 * (1.f / 32.f) - mu * mu;
  float xn  = (x - mu) * rsqrtf(var + 1e-3f) + beta[c];
  Ygn[bs * DIMC + c] = (__bf16)xn;
}

// ------------------------------------------- Wgate = swish(q@Wg + bg) (f32)
// 32x64 tile per wave: every B-fragment feeds two WMMAs (halves B traffic).
__global__ __launch_bounds__(256) void gate_gemm_kernel(const __bf16* qbf, const __bf16* WgT,
                                                        const float* bg, float* Wgate) {
  int wave = threadIdx.x >> 5, lane = threadIdx.x & 31;
  int gtile = blockIdx.x * 8 + wave;            // 256 m-blocks x 16 n-groups
  int m0 = (gtile >> 4) * 32, n0 = (gtile & 15) * 64;
  v8f acc[2][4];
#pragma unroll
  for (int mi = 0; mi < 2; ++mi)
#pragma unroll
    for (int j = 0; j < 4; ++j) acc[mi][j] = v8f_zero();
  for (int k0 = 0; k0 < DIMC; k0 += 32) {
    Frag a0, a1;
    load_a_frag(qbf, DIMC, m0,      k0, lane, &a0);
    load_a_frag(qbf, DIMC, m0 + 16, k0, lane, &a1);
    if (k0 + 64 < DIMC)   // hint the streaming operand ahead (global_prefetch_b8)
      __builtin_prefetch(qbf + (size_t)(m0 + (lane & 15)) * DIMC + k0 + 64, 0, 1);
#pragma unroll
    for (int j = 0; j < 4; ++j) {
      Frag bfr;
      load_b_frag(WgT, DIMC, n0 + j * 16, k0, lane, &bfr);
      acc[0][j] = wmma_bf16(a0, bfr, acc[0][j]);
      acc[1][j] = wmma_bf16(a1, bfr, acc[1][j]);
    }
  }
  int n = lane & 15, hi = lane >> 4;
#pragma unroll
  for (int mi = 0; mi < 2; ++mi)
#pragma unroll
    for (int j = 0; j < 4; ++j) {
      int col = n0 + j * 16 + n;
      float bias = bg[col];
#pragma unroll
      for (int r = 0; r < 8; ++r) {
        int m = mi * 16 + r + 8 * hi;
        float x = acc[mi][j][r] + bias;
        Wgate[(size_t)(m0 + m) * DIMC + col] = x / (1.f + expf(-x));
      }
    }
}

// -------------------------------- out = Wgate * (GN(Y)@Wo) + bo  (f32 out)
__global__ __launch_bounds__(256) void out_gemm_kernel(const __bf16* Ygn, const __bf16* WoT,
                                                       const float* bo, const float* Wgate,
                                                       float* outp) {
  int wave = threadIdx.x >> 5, lane = threadIdx.x & 31;
  int gtile = blockIdx.x * 8 + wave;
  int m0 = (gtile >> 4) * 32, n0 = (gtile & 15) * 64;
  v8f acc[2][4];
#pragma unroll
  for (int mi = 0; mi < 2; ++mi)
#pragma unroll
    for (int j = 0; j < 4; ++j) acc[mi][j] = v8f_zero();
  for (int k0 = 0; k0 < DIMC; k0 += 32) {
    Frag a0, a1;
    load_a_frag(Ygn, DIMC, m0,      k0, lane, &a0);
    load_a_frag(Ygn, DIMC, m0 + 16, k0, lane, &a1);
    if (k0 + 64 < DIMC)
      __builtin_prefetch(Ygn + (size_t)(m0 + (lane & 15)) * DIMC + k0 + 64, 0, 1);
#pragma unroll
    for (int j = 0; j < 4; ++j) {
      Frag bfr;
      load_b_frag(WoT, DIMC, n0 + j * 16, k0, lane, &bfr);
      acc[0][j] = wmma_bf16(a0, bfr, acc[0][j]);
      acc[1][j] = wmma_bf16(a1, bfr, acc[1][j]);
    }
  }
  int n = lane & 15, hi = lane >> 4;
#pragma unroll
  for (int mi = 0; mi < 2; ++mi)
#pragma unroll
    for (int j = 0; j < 4; ++j) {
      int col = n0 + j * 16 + n;
      float bias = bo[col];
#pragma unroll
      for (int r = 0; r < 8; ++r) {
        int m = mi * 16 + r + 8 * hi;
        size_t idx = (size_t)(m0 + m) * DIMC + col;
        outp[idx] = Wgate[idx] * acc[mi][j][r] + bias;
      }
    }
}

// ---------------------------------------------------------------- launcher
extern "C" void kernel_launch(void* const* d_in, const int* in_sizes, int n_in,
                              void* d_out, int out_size, void* d_ws, size_t ws_size,
                              hipStream_t stream) {
  const float* q      = (const float*)d_in[0];
  // d_in[1] (k) and d_in[2] (v) are dead in the reference (it uses qh for all three)
  const float* W_qkv  = (const float*)d_in[3];
  const float* Wg     = (const float*)d_in[4];
  const float* bg     = (const float*)d_in[5];
  const float* Wo     = (const float*)d_in[6];
  const float* bo     = (const float*)d_in[7];
  const float* beta   = (const float*)d_in[8];
  float* outp = (float*)d_out;

  char* ws = (char*)d_ws;
  size_t off = 0;
  auto alloc = [&](size_t bytes) -> void* {
    void* p = ws + off;
    off = (off + bytes + 255) & ~(size_t)255;
    return p;
  };
  __bf16* qbf    = (__bf16*)alloc((size_t)BB * SEQ * DIMC * 2);
  __bf16* WgT    = (__bf16*)alloc((size_t)DIMC * DIMC * 2);
  __bf16* WoT    = (__bf16*)alloc((size_t)DIMC * DIMC * 2);
  __bf16* WqkvT  = (__bf16*)alloc((size_t)NH * HD * HD * 2);
  __bf16* Xbf    = (__bf16*)alloc((size_t)BB * NH * SEQ * HD * 2);
  __bf16* XbfT   = (__bf16*)alloc((size_t)BB * NH * SEQ * HD * 2);
  float*  colsum = (float*) alloc((size_t)BB * NH * SEQ * 4);
  float*  Wgate  = (float*) alloc((size_t)BB * SEQ * DIMC * 4);
  float*  Y      = (float*) alloc((size_t)BB * SEQ * DIMC * 4);
  __bf16* Ygn    = (__bf16*)alloc((size_t)BB * SEQ * DIMC * 2);

  int nq = BB * SEQ * DIMC;
  cvt_bf16_kernel<<<(nq + 255) / 256, 256, 0, stream>>>(q, qbf, nq);
  transpose_cvt_kernel<<<dim3(DIMC / 16, DIMC / 16, 1), 256, 0, stream>>>(Wg, WgT, DIMC, DIMC);
  transpose_cvt_kernel<<<dim3(DIMC / 16, DIMC / 16, 1), 256, 0, stream>>>(Wo, WoT, DIMC, DIMC);
  transpose_cvt_kernel<<<dim3(HD / 16, HD / 16, NH), 256, 0, stream>>>(W_qkv, WqkvT, HD, HD);

  proj_kernel<<<dim3(SEQ / 16, NH, BB), 256, 0, stream>>>(qbf, WqkvT, Xbf, XbfT);
  gate_gemm_kernel<<<(256 * 16) / 8, 256, 0, stream>>>(qbf, WgT, bg, Wgate);

  pass1_colsum_kernel<<<dim3(SEQ / 128, NH, BB), 256, 16 * HD * 2, stream>>>(Xbf, colsum);
  pass2_out_kernel<<<dim3(SEQ / 16 / 8, NH, BB), 256, 0, stream>>>(Xbf, XbfT, colsum, Y);

  groupnorm_kernel<<<BB * SEQ, 1024, 0, stream>>>(Y, beta, Ygn);
  out_gemm_kernel<<<(256 * 16) / 8, 256, 0, stream>>>(Ygn, WoT, bo, Wgate, outp);
}
